// Attention_57681410785326
// MI455X (gfx1250) — matrix operand order repaired and checked by reading, not compiled, not run on hardware
//
#include <hip/hip_runtime.h>

// CDNA5 (gfx1250) wave32 paged-attention decode with bf16 WMMA.
typedef __attribute__((ext_vector_type(16))) __bf16 v16bf;
typedef __attribute__((ext_vector_type(8)))  float  v8f;
typedef __attribute__((ext_vector_type(4)))  float  v4f;

#define NB      32
#define KVLEN   2048
#define PAGE_SZ 16
#define NPAGES  128      // KVLEN / PAGE_SZ
#define HQ      32
#define HKV     8
#define GQ      4        // HQ / HKV
#define HD      128
#define NWAVES  8
#define KPW     (KVLEN / NWAVES)   // 256 keys per wave
#define ITER_K  32

// SCALE * log2(e): softmax done in exp2 domain.
#define SC_LOG2E (0.08838834764831845f * 1.4426950408889634f)

__device__ __forceinline__ const float* dec_ptr(unsigned e, const float* cache,
                                                const float* fresh) {
  return (e & 0x80000000u) ? (fresh + (e & 0x7fffffffu)) : (cache + e);
}

__global__ __launch_bounds__(256, 1)
void pa_decode_wmma(const float* __restrict__ q,   const float* __restrict__ kin,
                    const float* __restrict__ vin, const float* __restrict__ kc,
                    const float* __restrict__ vc,  const int* __restrict__ bt,
                    const int* __restrict__ ctx_lens, float* __restrict__ out)
{
  const int b    = (int)blockIdx.x >> 3;   // batch
  const int h    = (int)blockIdx.x & 7;    // kv head
  const int tid  = (int)threadIdx.x;
  const int wave = tid >> 5;
  const int lane = tid & 31;
  const int lo   = lane & 15;
  const bool up  = lane >= 16;
  const int ctx  = ctx_lens[b];

  __shared__ float part_O[NWAVES * GQ * HD];   // 16 KB
  __shared__ float part_m[NWAVES * GQ];
  __shared__ float part_s[NWAVES * GQ];

  // ---- Preload Q as WMMA B-operand (Q^T: K=dims, N=query-group g) ----
  // B layout: lanes 0-15 hold K=d0..d0+7,d0+16..d0+23 of column g=lane; lanes 16-31 hold
  // K=d0+8..d0+15,d0+24..d0+31 of column g=lane-16. Columns g>=4 are zero padding.
  v16bf qb[4];
  {
    const int g = lo;
    if (g < GQ) {
      const float* qrow = q + (size_t)((b * HQ + h * GQ + g)) * HD;
      #pragma unroll
      for (int st = 0; st < 4; ++st) {
        const int db = st * 32 + (up ? 8 : 0);
        v16bf t;
        #pragma unroll
        for (int i = 0; i < 16; ++i) t[i] = (__bf16)qrow[db + (i < 8 ? i : i + 8)];
        qb[st] = t;
      }
    } else {
      #pragma unroll
      for (int st = 0; st < 4; ++st) {
        v16bf t;
        #pragma unroll
        for (int i = 0; i < 16; ++i) t[i] = (__bf16)0.0f;
        qb[st] = t;
      }
    }
  }

  // Online softmax state: lane owns row g = lane&15 (dup across lane, lane+16).
  float m    = -__builtin_inff();
  float ssum = 0.0f;
  v8f acc[8] = {};   // O accumulators: acc[c][r] = O[g=r][16c + lane%16]

  const int start = wave * KPW;
  const int stop  = (start + KPW < ctx) ? (start + KPW) : ctx;   // uniform

  for (int base = start; base < stop; base += ITER_K) {
    // ---- Per-lane key metadata: lane <-> key (base+lane) ----
    const int pos   = base + lane;                       // < KVLEN always
    const int page  = bt[b * NPAGES + (pos >> 4)];
    const int slot  = (page << 4) | (pos & 15);
    const unsigned koff = ((unsigned)slot * HKV + (unsigned)h) * HD;
    const unsigned foff = ((unsigned)(b * HKV + h)) * HD;
    const bool fresh = (pos == ctx - 1);                 // new token lives in k/v inputs
    const unsigned enc = fresh ? (foff | 0x80000000u) : koff;

    // K row pointers for the two 16-key score tiles (A-operand rows).
    const unsigned ek0 = __shfl(enc, lo, 32);
    const unsigned ek1 = __shfl(enc, 16 + lo, 32);
    const float* kr0 = dec_ptr(ek0, kc, kin);
    const float* kr1 = dec_ptr(ek1, kc, kin);

    // V row pointers: lower lanes need keys 0..7,16..23, upper lanes keys 8..15,24..31.
    const float* vp[16];
    #pragma unroll
    for (int j = 0; j < 16; ++j) {
      const unsigned ev = __shfl(enc, (up ? 8 : 0) + (j < 8 ? j : j + 8), 32);
      vp[j] = dec_ptr(ev, vc, vin);
    }

    // ---- QK^T: S^T tiles via 8 bf16 WMMAs (K=32 dims per step) ----
    v8f s0 = {}, s1 = {};
    #pragma unroll
    for (int st = 0; st < 4; ++st) {
      const int d0   = st * 32;
      const int doff = d0 + (up ? 8 : 0);
      // A layout (16x32 bf16): lane holds its key's dims {doff..+7, doff+16..+23}
      v4f a0 = *(const v4f*)(kr0 + doff);
      v4f a1 = *(const v4f*)(kr0 + doff + 4);
      v4f a2 = *(const v4f*)(kr0 + doff + 16);
      v4f a3 = *(const v4f*)(kr0 + doff + 20);
      v16bf A0;
      #pragma unroll
      for (int i = 0; i < 4; ++i) {
        A0[i]      = (__bf16)a0[i];
        A0[4 + i]  = (__bf16)a1[i];
        A0[8 + i]  = (__bf16)a2[i];
        A0[12 + i] = (__bf16)a3[i];
      }
      s0 = __builtin_amdgcn_wmma_f32_16x16x32_bf16(false, A0, false, qb[st],
                                                   (short)0, s0, false, false);
      v4f b0 = *(const v4f*)(kr1 + doff);
      v4f b1 = *(const v4f*)(kr1 + doff + 4);
      v4f b2 = *(const v4f*)(kr1 + doff + 16);
      v4f b3 = *(const v4f*)(kr1 + doff + 20);
      v16bf A1;
      #pragma unroll
      for (int i = 0; i < 4; ++i) {
        A1[i]      = (__bf16)b0[i];
        A1[4 + i]  = (__bf16)b1[i];
        A1[8 + i]  = (__bf16)b2[i];
        A1[12 + i] = (__bf16)b3[i];
      }
      s1 = __builtin_amdgcn_wmma_f32_16x16x32_bf16(false, A1, false, qb[st],
                                                   (short)0, s1, false, false);
    }

    // ---- Mask + scale (exp2 domain). Lane's keys: rbase+r and rbase+16+r ----
    const int rbase = base + (up ? 8 : 0);
    float t0[8], t1[8];
    #pragma unroll
    for (int r = 0; r < 8; ++r) {
      t0[r] = (rbase + r      < ctx) ? s0[r] * SC_LOG2E : -__builtin_inff();
      t1[r] = (rbase + 16 + r < ctx) ? s1[r] * SC_LOG2E : -__builtin_inff();
    }

    // Row max across this lane's 16 keys, then combine with partner lane^16.
    float mt = t0[0];
    #pragma unroll
    for (int r = 1; r < 8; ++r) mt = fmaxf(mt, t0[r]);
    #pragma unroll
    for (int r = 0; r < 8; ++r) mt = fmaxf(mt, t1[r]);
    mt = fmaxf(mt, __shfl_xor(mt, 16, 32));
    const float mnew  = fmaxf(m, mt);
    const float alpha = exp2f(m - mnew);   // m==-inf -> 0 (acc is zero then)

    float p0[8], p1[8];
    float ls = 0.0f;
    #pragma unroll
    for (int r = 0; r < 8; ++r) {
      p0[r] = exp2f(t0[r] - mnew);
      p1[r] = exp2f(t1[r] - mnew);
      ls += p0[r] + p1[r];
    }
    ls += __shfl_xor(ls, 16, 32);
    ssum = ssum * alpha + ls;
    m    = mnew;

    // Rescale accumulators: row r needs alpha of g=r (broadcast from lane r).
    v8f av;
    #pragma unroll
    for (int r = 0; r < 8; ++r) av[r] = __shfl(alpha, r, 32);
    #pragma unroll
    for (int c = 0; c < 8; ++c) acc[c] = acc[c] * av;

    // ---- P is already in A-operand layout (M=g rows, K=32 keys): pack bf16 ----
    v16bf P;
    #pragma unroll
    for (int r = 0; r < 8; ++r) {
      P[r]     = (__bf16)p0[r];
      P[8 + r] = (__bf16)p1[r];
    }

    // ---- PV: O += P * V ; 8 WMMAs, one per 16-dim chunk of D=128 ----
    #pragma unroll
    for (int c = 0; c < 8; ++c) {
      const int d = 16 * c + lo;
      v16bf Bv;   // B layout: element s = V[key 8h+s, or 8h+s+8 if s>=8][d], h = lane/16 (coalesced 64B/load)
      #pragma unroll
      for (int j = 0; j < 8; ++j) {
        Bv[2 * j]     = (__bf16)vp[2 * j][d];
        Bv[2 * j + 1] = (__bf16)vp[2 * j + 1][d];
      }
      acc[c] = __builtin_amdgcn_wmma_f32_16x16x32_bf16(false, P, false, Bv,
                                                       (short)0, acc[c], false, false);
    }
  }

  // ---- Flash-decoding combine across the 8 waves via LDS ----
  if (lane < 16) {
    #pragma unroll
    for (int r = 0; r < GQ; ++r)
      #pragma unroll
      for (int c = 0; c < 8; ++c)
        part_O[(wave * GQ + r) * HD + 16 * c + lane] = acc[c][r];
  }
  if (lane < GQ) {
    part_m[wave * GQ + lane] = m;
    part_s[wave * GQ + lane] = ssum;
  }
  __syncthreads();

  for (int e = tid; e < GQ * HD; e += 256) {
    const int g = e >> 7;
    const int d = e & (HD - 1);
    float M = -__builtin_inff();
    #pragma unroll
    for (int w = 0; w < NWAVES; ++w) M = fmaxf(M, part_m[w * GQ + g]);
    const float Mv = (M == -__builtin_inff()) ? 0.0f : M;
    float S = 0.0f, O = 0.0f;
    #pragma unroll
    for (int w = 0; w < NWAVES; ++w) {
      const float wgt = exp2f(part_m[w * GQ + g] - Mv);
      S += wgt * part_s[w * GQ + g];
      O += wgt * part_O[(w * GQ + g) * HD + d];
    }
    out[(size_t)((b * HQ + h * GQ + g)) * HD + d] = (S > 0.0f) ? (O / S) : 0.0f;
  }
}

extern "C" void kernel_launch(void* const* d_in, const int* in_sizes, int n_in,
                              void* d_out, int out_size, void* d_ws, size_t ws_size,
                              hipStream_t stream) {
  const float* q    = (const float*)d_in[0];
  const float* k    = (const float*)d_in[1];
  const float* v    = (const float*)d_in[2];
  const float* kc   = (const float*)d_in[3];
  const float* vc   = (const float*)d_in[4];
  // d_in[5] = slot_mapping (unused: new token position inferred as ctx-1)
  const int* bt     = (const int*)d_in[6];
  const int* ctx    = (const int*)d_in[7];
  float* out        = (float*)d_out;

  dim3 grid(NB * HKV);   // 256 workgroups: one per (batch, kv-head)
  dim3 block(256);       // 8 wave32s, flash-decoding split over KV
  pa_decode_wmma<<<grid, block, 0, stream>>>(q, k, v, kc, vc, bt, ctx, out);
}
